// SubGraph_40664750359214
// MI455X (gfx1250) — matrix operand order repaired and checked by reading, not compiled, not run on hardware
//
#include <hip/hip_runtime.h>

#define EPSV 1e-5f

typedef __attribute__((ext_vector_type(16))) __bf16          v16bf;
typedef __attribute__((ext_vector_type(8)))  float           v8f;
typedef __attribute__((ext_vector_type(8)))  unsigned short  v8us;

union FragBF {
  v16bf v;
  v8us  h[2];
  unsigned short s[16];
};

__device__ __forceinline__ unsigned short f2bf_bits(float f) {
  unsigned u = __builtin_bit_cast(unsigned, f);
  unsigned r = u + 0x7FFFu + ((u >> 16) & 1u);   // round-to-nearest-even
  return (unsigned short)(r >> 16);
}
__device__ __forceinline__ float bf2f(unsigned short s) {
  return __builtin_bit_cast(float, ((unsigned)s) << 16);
}
// fused BN affine + relu applied on-the-fly from stored pre-BN bf16 activations
__device__ __forceinline__ float actv(unsigned short z, const float* coef, int c) {
  float v = coef[c] * bf2f(z) + coef[64 + c];
  return v > 0.f ? v : 0.f;
}

// ---------------------------------------------------------------------------
// WMMA GEMM:  Z[rows,64] = A[rows,KK] @ W[KK,64] + bias, with BN-stat
// accumulation.  A is assembled per MODE:
//  MODE 1: A = concat(x[rows_src], dsum[rows_dom])           (rows = R)
//  MODE 2: A = concat(relu(aff1(z1)), msg[rows_dom])         (rows = R)
//  MODE 3: A = concat(relu(aff2(z2[:N])), agg)               (rows = N)
//  MODE 4: A = relu(aff3(z3))  (KK=64), relu applied to out  (rows = N)
// Block: 256 threads = 8 waves; 32 rows x 64 cols; wave -> one 16x16 tile.
// LDS: A row-major (padded), B transposed (bt[n][k], padded) so every lane's
// fragment is two contiguous 16B runs -> ds_load_b128, conflict-free banks.
// ---------------------------------------------------------------------------
template <int MODE, int KK>
__global__ __launch_bounds__(256) void gemm_bn_kernel(
    const float* __restrict__ xsrc,          // MODE1: x
    const float* __restrict__ side,          // MODE1: dsum, MODE2: msg, MODE3: agg
    const unsigned short* __restrict__ zin,  // MODE2/3/4: previous z (bf16)
    const float* __restrict__ coef,          // previous BN affine [scale|shift]
    const int* __restrict__ src,
    const int* __restrict__ dst,
    const float* __restrict__ W,             // KK x 64 fp32 (row-major)
    const float* __restrict__ bias,          // 64
    unsigned short* __restrict__ zout,       // rows x 64 bf16
    float* __restrict__ stats,               // 128 floats: [sum | sumsq]
    int n_rows, int NN)
{
  constexpr int LST = KK + 8;                // padded LDS row stride (272B rows)
  __shared__ unsigned short a_s[32 * LST];   // A: [row][k]
  __shared__ unsigned short b_s[64 * LST];   // B transposed: [n][k]
  __shared__ float ssum[128];

  const int t = threadIdx.x;
  if (t < 128) ssum[t] = 0.f;

  // ---- stage W as bf16, transposed, in LDS: b_s[n*LST + k] = W[k][n]
  for (int idx = t; idx < KK * 64; idx += 256) {
    int k = idx >> 6, n = idx & 63;
    b_s[n * LST + k] = f2bf_bits(W[idx]);
  }

  // ---- assemble A tile (32 rows x KK), bf16 in LDS
  {
    const int r   = t >> 3;
    const int cc0 = (t & 7) * (KK / 8);
    const int gr  = blockIdx.x * 32 + r;
    const bool valid = gr < n_rows;
    int i1 = 0, i2 = 0;
    if (valid) {
      if (gr < NN) { i1 = gr; i2 = gr; }
      else {
        int e = gr - NN;
        i1 = src ? src[e] : 0;
        i2 = dst ? dst[e] : 0;
      }
    }
    #pragma unroll
    for (int j = 0; j < KK / 8; ++j) {
      int c = cc0 + j;
      float val = 0.f;
      if (valid) {
        if (MODE == 1)      val = (c < 64) ? xsrc[i1 * 64 + c] : side[i2 * 64 + (c - 64)];
        else if (MODE == 2) val = (c < 64) ? actv(zin[gr * 64 + c], coef, c)
                                           : side[i2 * 64 + (c - 64)];
        else if (MODE == 3) val = (c < 64) ? actv(zin[gr * 64 + c], coef, c)
                                           : side[gr * 64 + (c - 64)];
        else                val = actv(zin[gr * 64 + c], coef, c);
      }
      a_s[r * LST + c] = f2bf_bits(val);
    }
  }
  __syncthreads();

  const int wid  = t >> 5;
  const int lane = t & 31;
  const int mt = wid >> 2;        // 0..1 : M tile
  const int nt = wid & 3;         // 0..3 : N tile
  const int lh = lane >> 4;       // lane half
  const int lm = lane & 15;
  const int m0 = mt * 16;
  const int n  = nt * 16 + lm;    // output channel for this lane

  const unsigned short* ap = &a_s[(m0 + lm) * LST];
  const unsigned short* bp = &b_s[n * LST];

  v8f acc = {0.f, 0.f, 0.f, 0.f, 0.f, 0.f, 0.f, 0.f};
  #pragma unroll
  for (int ks = 0; ks < KK / 32; ++ks) {
    FragBF fa, fb;
    // A-frag (ISA 7.12.2, 16-bit A 16x32): lane half lh holds K = lh*8 + {0..7}
    // and K = 16 + lh*8 + {0..7} -> two contiguous 16B runs.
    fa.h[0] = *(const v8us*)(ap + ks * 32 + lh * 8);
    fa.h[1] = *(const v8us*)(ap + ks * 32 + 16 + lh * 8);
    // B-frag: lane half lh holds K = lh*8 + {0..7} and K = 16 + lh*8 + {0..7} of column n.
    fb.h[0] = *(const v8us*)(bp + ks * 32 + lh * 8);
    fb.h[1] = *(const v8us*)(bp + ks * 32 + 16 + lh * 8);
    acc = __builtin_amdgcn_wmma_f32_16x16x32_bf16(false, fa.v, false, fb.v,
                                                  (short)0, acc, false, false);
  }

  // ---- bias (+relu for MODE4), store bf16, accumulate BN stats.
  // Fast path: whole 32-row tile valid (every block except the ragged tail)
  // -> straight-line epilogue with no per-row exec-mask branching.
  const float bn = bias[n];
  const int grb = blockIdx.x * 32 + m0 + lh * 8;  // first row this lane touches
  float ps = 0.f, pq = 0.f;
  if (blockIdx.x * 32 + 32 <= n_rows) {
    #pragma unroll
    for (int v = 0; v < 8; ++v) {
      float val = acc[v] + bn;                     // C/D map: VGPR v -> row m0+v(+8)
      if (MODE == 4) val = val > 0.f ? val : 0.f;  // stats post-relu in last layer
      zout[(grb + v) * 64 + n] = f2bf_bits(val);
      ps += val;
      pq += val * val;
    }
  } else {
    #pragma unroll
    for (int v = 0; v < 8; ++v) {
      if (grb + v < n_rows) {
        float val = acc[v] + bn;
        if (MODE == 4) val = val > 0.f ? val : 0.f;
        zout[(grb + v) * 64 + n] = f2bf_bits(val);
        ps += val;
        pq += val * val;
      }
    }
  }
  atomicAdd(&ssum[n], ps);
  atomicAdd(&ssum[64 + n], pq);
  __syncthreads();
  if (t < 128) atomicAdd(&stats[t], ssum[t]);
}

// ---------------------------------------------------------------------------
// Helper kernels
// ---------------------------------------------------------------------------
__global__ void k_zero_f(float* p, int n) {
  int i = blockIdx.x * blockDim.x + threadIdx.x;
  if (i < n) p[i] = 0.f;
}

__global__ void k_copy_f(float* d, const float* s, int n) {
  int i = blockIdx.x * blockDim.x + threadIdx.x;
  if (i < n) d[i] = s[i];
}

// accum[sc[e], c] += vals[g[e], c]   (64 channels per edge, f32 L2 atomics)
__global__ void k_scatter_plain(float* __restrict__ acc, const float* __restrict__ vals,
                                const int* __restrict__ g, const int* __restrict__ sc, int E) {
  int t = blockIdx.x * blockDim.x + threadIdx.x;
  if (t >= E * 64) return;
  int e = t >> 6, c = t & 63;
  atomicAdd(&acc[sc[e] * 64 + c], vals[g[e] * 64 + c]);
}

// acc[i] = relu(affine(z_self[i]))  -- self-row initialization of a segment sum
__global__ void k_init_act(float* __restrict__ acc, const unsigned short* __restrict__ zin,
                           const float* __restrict__ coef, int NN) {
  int t = blockIdx.x * blockDim.x + threadIdx.x;
  if (t >= NN * 64) return;
  acc[t] = actv(zin[t], coef, t & 63);
}

// acc[sc[e], c] += relu(affine(z_edge[e, c]))
__global__ void k_scatter_act(float* __restrict__ acc, const unsigned short* __restrict__ zin,
                              const float* __restrict__ coef, const int* __restrict__ sc,
                              int NN, int E) {
  int t = blockIdx.x * blockDim.x + threadIdx.x;
  if (t >= E * 64) return;
  int e = t >> 6, c = t & 63;
  atomicAdd(&acc[sc[e] * 64 + c], actv(zin[(NN + e) * 64 + c], coef, c));
}

// fold BN (mean/var from stats) + gamma/beta into per-channel scale/shift
__global__ void k_bn_coef(const float* __restrict__ stats, float invcnt,
                          const float* __restrict__ gamma, const float* __restrict__ beta,
                          float* __restrict__ coef) {
  int c = threadIdx.x;
  if (c >= 64) return;
  float mean = stats[c] * invcnt;
  float var  = stats[64 + c] * invcnt - mean * mean;
  float sc   = gamma[c] * rsqrtf(var + EPSV);
  coef[c]      = sc;
  coef[64 + c] = beta[c] - mean * sc;
}

__global__ void k_final_out(float* __restrict__ out, const unsigned short* __restrict__ u,
                            const float* __restrict__ coef, int NN) {
  int t = blockIdx.x * blockDim.x + threadIdx.x;
  if (t >= NN * 64) return;
  out[t] = actv(u[t], coef, t & 63);
}

// ---------------------------------------------------------------------------
extern "C" void kernel_launch(void* const* d_in, const int* in_sizes, int n_in,
                              void* d_out, int out_size, void* d_ws, size_t ws_size,
                              hipStream_t stream) {
  const float* x          = (const float*)d_in[0];
  const int*   ei         = (const int*)  d_in[1];
  const float* W_unite    = (const float*)d_in[2];
  const float* b_unite    = (const float*)d_in[3];
  const float* g1         = (const float*)d_in[4];
  const float* be1        = (const float*)d_in[5];
  const float* W_gconv    = (const float*)d_in[6];
  const float* b_gconv    = (const float*)d_in[7];
  const float* g2         = (const float*)d_in[8];
  const float* be2        = (const float*)d_in[9];
  const float* W_transfer = (const float*)d_in[10];
  const float* b_transfer = (const float*)d_in[11];
  const float* W_lin      = (const float*)d_in[12];
  const float* b_lin      = (const float*)d_in[13];
  const float* g3         = (const float*)d_in[14];
  const float* be3        = (const float*)d_in[15];

  const int N = in_sizes[0] / 64;
  const int E = in_sizes[1] / 2;
  const int R = N + E;
  const int* src = ei;
  const int* dst = ei + E;

  char* ws = (char*)d_ws;
  size_t off = 0;
  auto take = [&](size_t bytes) -> char* {
    char* p = ws + off;
    off += (bytes + 255) & ~(size_t)255;
    return p;
  };

  float* dsum  = (float*)take((size_t)N * 64 * 4);
  float* s_acc = (float*)take((size_t)N * 64 * 4);
  float* msg   = (float*)take((size_t)N * 64 * 4);
  float* agg   = (float*)take((size_t)N * 64 * 4);
  unsigned short* z1 = (unsigned short*)take((size_t)R * 64 * 2);
  unsigned short* z2 = (unsigned short*)take((size_t)R * 64 * 2);
  unsigned short* z3 = (unsigned short*)take((size_t)N * 64 * 2);
  unsigned short* uu = (unsigned short*)take((size_t)N * 64 * 2);
  float* stats = (float*)take(4 * 128 * 4);
  float* coef  = (float*)take(4 * 128 * 4);

  const int TB = 256;
  const int gN  = (N * 64 + TB - 1) / TB;   // node-row element grid
  const int gE  = (E * 64 + TB - 1) / TB;   // edge-row element grid
  const int gbR = (R + 31) / 32;            // GEMM blocks over R rows
  const int gbN = (N + 31) / 32;            // GEMM blocks over N rows

  // fresh stats every call (atomically accumulated below)
  k_zero_f<<<(512 + TB - 1) / TB, TB, 0, stream>>>(stats, 512);

  // ---- dsum = segment_sum(x[rows_src], rows_dom)
  k_copy_f<<<gN, TB, 0, stream>>>(dsum, x, N * 64);
  k_scatter_plain<<<gE, TB, 0, stream>>>(dsum, x, src, dst, E);

  // ---- layer 1: z1 = concat(f, dsum[dom]) @ W_unite + b  (stats -> BN1)
  gemm_bn_kernel<1, 128><<<gbR, TB, 0, stream>>>(x, dsum, nullptr, nullptr, src, dst,
                                                 W_unite, b_unite, z1, stats + 0, R, N);
  k_bn_coef<<<1, 64, 0, stream>>>(stats + 0, 1.f / (float)R, g1, be1, coef + 0);

  // ---- s = segment_sum(h, rows_dom); msg = segment_sum(s[src], dst)
  k_init_act<<<gN, TB, 0, stream>>>(s_acc, z1, coef + 0, N);
  k_scatter_act<<<gE, TB, 0, stream>>>(s_acc, z1, coef + 0, dst, N, E);
  k_zero_f<<<gN, TB, 0, stream>>>(msg, N * 64);
  k_scatter_plain<<<gE, TB, 0, stream>>>(msg, s_acc, src, dst, E);

  // ---- layer 2: z2 = concat(h, msg[dom]) @ W_gconv + b  (stats -> BN2)
  gemm_bn_kernel<2, 128><<<gbR, TB, 0, stream>>>(nullptr, msg, z1, coef + 0, src, dst,
                                                 W_gconv, b_gconv, z2, stats + 128, R, N);
  k_bn_coef<<<1, 64, 0, stream>>>(stats + 128, 1.f / (float)R, g2, be2, coef + 128);

  // ---- agg = segment_sum(h2, rows_src)
  k_init_act<<<gN, TB, 0, stream>>>(agg, z2, coef + 128, N);
  k_scatter_act<<<gE, TB, 0, stream>>>(agg, z2, coef + 128, src, N, E);

  // ---- layer 3: z3 = concat(h2[:N], agg) @ W_transfer + b  (BN reuses g2/be2)
  gemm_bn_kernel<3, 128><<<gbN, TB, 0, stream>>>(nullptr, agg, z2, coef + 128,
                                                 nullptr, nullptr, W_transfer, b_transfer,
                                                 z3, stats + 256, N, N);
  k_bn_coef<<<1, 64, 0, stream>>>(stats + 256, 1.f / (float)N, g2, be2, coef + 256);

  // ---- layer 4: u = relu(t @ W_lin + b); stats on post-relu u
  gemm_bn_kernel<4, 64><<<gbN, TB, 0, stream>>>(nullptr, nullptr, z3, coef + 256,
                                                nullptr, nullptr, W_lin, b_lin,
                                                uu, stats + 384, N, N);
  k_bn_coef<<<1, 64, 0, stream>>>(stats + 384, 1.f / (float)N, g3, be3, coef + 384);

  // ---- out = relu(BN(u))
  k_final_out<<<gN, TB, 0, stream>>>((float*)d_out, uu, coef + 384, N);
}
